// Edge_Sheaf_NNet_36206574305736
// MI455X (gfx1250) — matrix operand order repaired and checked by reading, not compiled, not run on hardware
//
#include <hip/hip_runtime.h>
#include <hip/hip_bf16.h>
#include <math.h>

// ---------------------------------------------------------------------------
// Problem constants (from the reference)
// ---------------------------------------------------------------------------
#define NNODE   20000
#define DX      16
#define DS      64
#define NLAB    10
#define NSMAT   64
#define EHALF   160000
#define NEDGE   (2 * EHALF)
#define NLABELED 2000
#define NHID    6

// Workspace layout (bytes):
//   [0, 98304)            : bf16 weights in B-fragment order (96 frags * 1KB)
//   [98304, 1378304)      : new_x  (N*16 f32)   -- zeroed per call
//   [1378304, 1458304)    : deg    (N f32)      -- zeroed per call
//   [1458304, 1458320)    : acc[4] f32          -- zeroed per call
//   [1458320, 2738320)    : xmaped (N*16 f32)
// Total ~2.75 MB (assumed <= ws_size).
#define WS_WB      0
#define WS_NEWX    98304
#define WS_DEG     1378304
#define WS_ACC     1458304
#define WS_XMAP    1458320

typedef __attribute__((ext_vector_type(16))) __bf16 v16bf;
typedef __attribute__((ext_vector_type(8)))  float  v8f;

union B16Frag { v16bf v; uint4 q[2]; };

__device__ inline v8f wmma_bf16(v16bf a, v16bf b, v8f c) {
  // D = A(16x32 bf16) x B(32x16 bf16) + C(16x16 f32)
  return __builtin_amdgcn_wmma_f32_16x16x32_bf16(
      /*neg_a=*/false, a, /*neg_b=*/false, b,
      /*c_mod=*/(short)0, c, /*reuse_a=*/false, /*reuse_b=*/false);
}

// A fragment (16x32 bf16) from a row-major LDS tile.
// lane: m = lane&15, kbase = (lane>>4)*8
// halves 0..7  <- K = off + kbase .. +7
// halves 8..15 <- K = off + kbase + 16 .. +23
__device__ inline v16bf afrag_lds(const __bf16* row, int off) {
  B16Frag u;
  u.q[0] = *(const uint4*)(row + off);
  u.q[1] = *(const uint4*)(row + off + 16);
  return u.v;
}

// B fragment (32x16 bf16) from the pre-swizzled global weight buffer.
// Each fragment is 32 lanes * 16 halves contiguous (1 KB).
__device__ inline v16bf bfrag_g(const __bf16* wB, int fragId, int lane) {
  const uint4* p = (const uint4*)(wB + (fragId << 9) + (lane << 4));
  B16Frag u; u.q[0] = p[0]; u.q[1] = p[1];
  return u.v;
}

// ---------------------------------------------------------------------------
// Weight pre-swizzle: fp32 row-major -> bf16 B-fragment order.
// Fragment ids: layer0: nt*4+kt (0..15); hidden li: 16+li*8+nt*2+kt (16..63);
// output: 64+nt*2+kt (64..95).
// B layout per fragment: lane l holds column n = l&15; half h <-> K = (l>>4)*8 + (h&7) + (h>>3)*16
// (plus ktile offset kt*32).
// ---------------------------------------------------------------------------
__global__ __launch_bounds__(256) void prep_weights_kernel(
    const float* __restrict__ fcW_in, const float* __restrict__ fcW_h,
    const float* __restrict__ fcW_out, unsigned short* __restrict__ wB_raw) {
  int t = blockIdx.x * blockDim.x + threadIdx.x;
  if (t >= 96 * 32) return;
  int frag = t >> 5, lane = t & 31;
  const float* W; int ncols, kt, nt;
  if (frag < 16) {
    kt = frag & 3; nt = frag >> 2; W = fcW_in; ncols = NSMAT;          // 128x64
  } else if (frag < 64) {
    int h = frag - 16; int li = h >> 3; int r = h & 7;
    kt = r & 1; nt = r >> 1; W = fcW_h + li * NSMAT * NSMAT; ncols = NSMAT;  // 64x64
  } else {
    int r = frag - 64; kt = r & 1; nt = r >> 1; W = fcW_out; ncols = DX * DX; // 64x256
  }
  int col = nt * 16 + (lane & 15);
  int k0  = kt * 32 + (lane >> 4) * 8;
  __bf16* outp = (__bf16*)wB_raw + frag * 512 + lane * 16;
#pragma unroll
  for (int h = 0; h < 16; ++h) outp[h] = (__bf16)W[(k0 + (h & 7) + (h >> 3) * 16) * ncols + col];
}

__global__ void init_zero_kernel(float* __restrict__ p, int n) {
  int i = blockIdx.x * blockDim.x + threadIdx.x;
  if (i < n) p[i] = 0.0f;
}

// ---------------------------------------------------------------------------
// Main kernel: one workgroup (2 wave32s) per 16 reverse-edge pairs.
//   wave 0: MLP for forward edges  g*16 .. g*16+15
//   wave 1: MLP for reverse edges  g*16+EHALF .. +15
// Then: message scatter (atomics) + 16 pair products A@B-I -> loss_orth.
// ---------------------------------------------------------------------------
__global__ __launch_bounds__(64) void sheaf_main_kernel(
    const float* __restrict__ xembed, const float* __restrict__ sembed,
    const float* __restrict__ edge_w,
    const float* __restrict__ fcb_in, const float* __restrict__ fcb_h,
    const float* __restrict__ fcb_out,
    const int* __restrict__ src, const int* __restrict__ tgt,
    const unsigned short* __restrict__ wB_raw,
    float* __restrict__ new_x, float* __restrict__ deg, float* __restrict__ acc) {
  __shared__ __align__(16) __bf16 sIn[2][16][128];   // input tiles (bf16)      8 KB
  __shared__ __align__(16) __bf16 sH[2][16][NSMAT];  // hidden activations      4 KB
  __shared__ __align__(16) float  sMat[32][256];     // edge matrices (f32)    32 KB
  __shared__ float  sX[32][DX];                      // xembed[src] rows        2 KB
  __shared__ int    sTgt[32];
  __shared__ float  sEw[32];

  const __bf16* wB = (const __bf16*)wB_raw;
  const int wave = threadIdx.x >> 5;
  const int lane = threadIdx.x & 31;
  const int g = blockIdx.x;
  const int eBase = g * 16 + wave * EHALF;

  const int mrow = lane & 15;          // A-frag / C-frag column index helpers
  const int kb   = (lane >> 4) * 8;
  const int hi   = (lane >> 4) * 8;    // C/D: m = r + 8*(lane>>4)
  const int ncol = lane & 15;

  // ---- Stage 1: gathers ----------------------------------------------------
  if (lane < 16) {
    int e = eBase + lane;
    sTgt[wave * 16 + lane] = tgt[e];
    sEw[wave * 16 + lane]  = edge_w[e];
  }
  // input features: 16 edges x 128 (sembed[src] || sembed[tgt]) as bf16
  for (int m = 0; m < 16; ++m) {
    int e = eBase + m;
    int s = src[e], tg = tgt[e];
#pragma unroll
    for (int it = 0; it < 4; ++it) {
      int k = lane + 32 * it;
      float f = (it < 2) ? sembed[s * DS + k] : sembed[tg * DS + (k - DS)];
      sIn[wave][m][k] = (__bf16)f;
    }
  }
  // xembed[src] rows for this wave's 16 edges
#pragma unroll
  for (int it = 0; it < 8; ++it) {
    int o = lane + 32 * it;            // 0..255
    int le = o >> 4, j = o & 15;
    int s = src[eBase + le];
    sX[wave * 16 + le][j] = xembed[s * DX + j];
  }
  __syncthreads();

  // ---- Stage 2: 8-layer MLP via WMMA --------------------------------------
  // Layer 0: [16x128] @ [128x64] + b, ReLU
  v16bf A4[4];
#pragma unroll
  for (int kt = 0; kt < 4; ++kt)
    A4[kt] = afrag_lds(&sIn[wave][mrow][0], kt * 32 + kb);

  v8f Hacc[4];
#pragma unroll
  for (int nt = 0; nt < 4; ++nt) {
    v8f c = {};
#pragma unroll
    for (int kt = 0; kt < 4; ++kt)
      c = wmma_bf16(A4[kt], bfrag_g(wB, nt * 4 + kt, lane), c);
    float b = fcb_in[nt * 16 + ncol];
#pragma unroll
    for (int r = 0; r < 8; ++r) { float x = c[r] + b; c[r] = x > 0.f ? x : 0.f; }
    Hacc[nt] = c;
  }
#pragma unroll
  for (int nt = 0; nt < 4; ++nt)
#pragma unroll
    for (int r = 0; r < 8; ++r)
      sH[wave][r + hi][nt * 16 + ncol] = (__bf16)Hacc[nt][r];

  // Hidden layers: 6 x ([16x64] @ [64x64] + b, ReLU)
#pragma unroll
  for (int li = 0; li < NHID; ++li) {
    v16bf A2[2];
#pragma unroll
    for (int kt = 0; kt < 2; ++kt)
      A2[kt] = afrag_lds(&sH[wave][mrow][0], kt * 32 + kb);
#pragma unroll
    for (int nt = 0; nt < 4; ++nt) {
      v8f c = {};
#pragma unroll
      for (int kt = 0; kt < 2; ++kt)
        c = wmma_bf16(A2[kt], bfrag_g(wB, 16 + li * 8 + nt * 2 + kt, lane), c);
      float b = fcb_h[li * NSMAT + nt * 16 + ncol];
#pragma unroll
      for (int r = 0; r < 8; ++r) { float x = c[r] + b; c[r] = x > 0.f ? x : 0.f; }
      Hacc[nt] = c;
    }
#pragma unroll
    for (int nt = 0; nt < 4; ++nt)
#pragma unroll
      for (int r = 0; r < 8; ++r)
        sH[wave][r + hi][nt * 16 + ncol] = (__bf16)Hacc[nt][r];
  }

  // Output layer: [16x64] @ [64x256] + b (no ReLU) -> sMat (f32)
  // ntile nt holds matrix row i=nt, columns j=0..15 for all 16 edges.
  {
    v16bf A2[2];
#pragma unroll
    for (int kt = 0; kt < 2; ++kt)
      A2[kt] = afrag_lds(&sH[wave][mrow][0], kt * 32 + kb);
#pragma unroll
    for (int nt = 0; nt < 16; ++nt) {
      v8f c = {};
#pragma unroll
      for (int kt = 0; kt < 2; ++kt)
        c = wmma_bf16(A2[kt], bfrag_g(wB, 64 + nt * 2 + kt, lane), c);
      float b = fcb_out[nt * 16 + ncol];
#pragma unroll
      for (int r = 0; r < 8; ++r)
        sMat[wave * 16 + r + hi][nt * 16 + ncol] = c[r] + b;
    }
  }
  __syncthreads();

  // ---- Stage 3: message passing (mat @ x_src, weighted scatter-add) -------
#pragma unroll
  for (int rep = 0; rep < 8; ++rep) {
    int o = threadIdx.x + 64 * rep;    // 0..511 = 32 edges x 16 outputs
    int le = o >> 4, i = o & 15;
    float s = 0.f;
#pragma unroll
    for (int j = 0; j < DX; ++j) s += sMat[le][i * DX + j] * sX[le][j];
    float w = sEw[le];
    atomicAdd(&new_x[(size_t)sTgt[le] * DX + i], w * s);
    if (i == 0) atomicAdd(&deg[sTgt[le]], w);
  }

  // ---- Stage 4: pair products A@B - I -> Frobenius norms -> loss_orth -----
  float orth = 0.f;
  for (int q = 0; q < 8; ++q) {
    int j = wave * 8 + q;              // pair index within block (0..15)
    // A = forward matrix, placed in K=0..15 of a 16x32 A-frag (zero-padded)
    v16bf a;
#pragma unroll
    for (int h = 0; h < 16; ++h) a[h] = (lane < 16) ? (__bf16)sMat[j][mrow * 16 + h] : (__bf16)0.0f;
    // B = reverse matrix: lanes 0-15 hold K=0..15 of column n; lanes 16-31 zero
    v16bf b;
    if (lane < 16) {
#pragma unroll
      for (int h = 0; h < 16; ++h) b[h] = (__bf16)sMat[16 + j][h * 16 + ncol];
    } else {
#pragma unroll
      for (int h = 0; h < 16; ++h) b[h] = (__bf16)0.0f;
    }
    v8f z = {};
    v8f p = wmma_bf16(a, b, z);        // uniform control flow: EXEC all ones
    float ss = 0.f;
#pragma unroll
    for (int r = 0; r < 8; ++r) {
      int mm = r + hi;
      float d = p[r] - ((mm == ncol) ? 1.0f : 0.0f);
      ss += d * d;
    }
#pragma unroll
    for (int off = 16; off > 0; off >>= 1) ss += __shfl_xor(ss, off, 32);
    orth += sqrtf(ss);                 // mean(diff^2)*(DX*DX) then sqrt == ||.||_F
  }
  if (lane == 0) atomicAdd(&acc[0], orth);
}

// ---------------------------------------------------------------------------
// Degree-normalize + loss_smap partial sums; store xmaped for the KL kernel.
// ---------------------------------------------------------------------------
__global__ __launch_bounds__(256) void xmap_kernel(
    const float* __restrict__ new_x, const float* __restrict__ deg,
    const float* __restrict__ xembed, float* __restrict__ xmaped,
    float* __restrict__ acc) {
  int idx = blockIdx.x * blockDim.x + threadIdx.x;
  float ss = 0.f;
  if (idx < NNODE * DX) {
    float v = new_x[idx];
    float d = deg[idx >> 4];
    float xm = (d > 0.f) ? v / fmaxf(d, 1e-20f) : v;
    xmaped[idx] = xm;
    float df = xm - xembed[idx];
    ss = df * df;
  }
#pragma unroll
  for (int off = 16; off > 0; off >>= 1) ss += __shfl_xor(ss, off, 32);
  if ((threadIdx.x & 31) == 0) atomicAdd(&acc[1], ss);
}

// ---------------------------------------------------------------------------
// KL(y || g) on labeled nodes: Linear(16->10) + LogSoftmax.
// ---------------------------------------------------------------------------
__global__ __launch_bounds__(128) void lbpr_kernel(
    const float* __restrict__ xmaped, const float* __restrict__ ylprob,
    const float* __restrict__ clW, const float* __restrict__ clb,
    const int* __restrict__ idvert, float* __restrict__ acc) {
  int t = blockIdx.x * blockDim.x + threadIdx.x;
  float kl = 0.f;
  if (t < NLABELED) {
    int node = idvert[t];
    float x[DX];
#pragma unroll
    for (int i = 0; i < DX; ++i) x[i] = xmaped[node * DX + i];
    float lg[NLAB];
#pragma unroll
    for (int l = 0; l < NLAB; ++l) {
      float s = clb[l];
#pragma unroll
      for (int i = 0; i < DX; ++i) s += x[i] * clW[i * NLAB + l];
      lg[l] = s;
    }
    float mx = lg[0];
#pragma unroll
    for (int l = 1; l < NLAB; ++l) mx = fmaxf(mx, lg[l]);
    float se = 0.f;
#pragma unroll
    for (int l = 0; l < NLAB; ++l) se += expf(lg[l] - mx);
    float lse = mx + logf(se);
#pragma unroll
    for (int l = 0; l < NLAB; ++l) {
      float ylp = ylprob[node * NLAB + l];
      kl += expf(ylp) * (ylp - (lg[l] - lse));
    }
  }
#pragma unroll
  for (int off = 16; off > 0; off >>= 1) kl += __shfl_xor(kl, off, 32);
  if ((threadIdx.x & 31) == 0) atomicAdd(&acc[2], kl);
}

__global__ void finalize_kernel(const float* __restrict__ acc, float* __restrict__ out) {
  if (threadIdx.x == 0 && blockIdx.x == 0) {
    out[0] = acc[0] / (float)EHALF;      // loss_orth
    out[1] = 0.0f;                       // loss_cons (exactly zero in reference)
    out[2] = acc[1] / (float)NNODE;      // loss_smap = mean(diff^2)*DX = sum/N
    out[3] = acc[2] / (float)NLABELED;   // loss_lbpr
  }
}

// ---------------------------------------------------------------------------
extern "C" void kernel_launch(void* const* d_in, const int* in_sizes, int n_in,
                              void* d_out, int out_size, void* d_ws, size_t ws_size,
                              hipStream_t stream) {
  const float* xembed  = (const float*)d_in[0];
  const float* sembed  = (const float*)d_in[1];
  const float* ylprob  = (const float*)d_in[2];
  const float* edge_w  = (const float*)d_in[3];
  const float* fcW_in  = (const float*)d_in[4];
  const float* fcb_in  = (const float*)d_in[5];
  const float* fcW_h   = (const float*)d_in[6];
  const float* fcb_h   = (const float*)d_in[7];
  const float* fcW_out = (const float*)d_in[8];
  const float* fcb_out = (const float*)d_in[9];
  const float* clW     = (const float*)d_in[10];
  const float* clb     = (const float*)d_in[11];
  const int*   src     = (const int*)d_in[12];
  const int*   tgt     = (const int*)d_in[13];
  // d_in[14]=rev1 (arange), d_in[15]=rev2 (arange+EHALF): structure exploited directly.
  const int*   idvert  = (const int*)d_in[16];
  (void)in_sizes; (void)n_in; (void)out_size; (void)ws_size;

  char* ws = (char*)d_ws;
  unsigned short* wB = (unsigned short*)(ws + WS_WB);
  float* new_x  = (float*)(ws + WS_NEWX);
  float* degp   = (float*)(ws + WS_DEG);
  float* accp   = (float*)(ws + WS_ACC);
  float* xmaped = (float*)(ws + WS_XMAP);

  // 1) weights -> bf16 B-fragment order
  prep_weights_kernel<<<12, 256, 0, stream>>>(fcW_in, fcW_h, fcW_out, wB);
  // 2) zero accumulators (new_x | deg | acc are contiguous: 320000+20000+4 floats)
  {
    int nzero = NNODE * DX + NNODE + 4;
    init_zero_kernel<<<(nzero + 255) / 256, 256, 0, stream>>>(new_x, nzero);
  }
  // 3) fused edge MLP + messages + pair products (WMMA path)
  sheaf_main_kernel<<<EHALF / 16, 64, 0, stream>>>(
      xembed, sembed, edge_w, fcb_in, fcb_h, fcb_out, src, tgt,
      wB, new_x, degp, accp);
  // 4) degree-normalize + loss_smap
  xmap_kernel<<<(NNODE * DX + 255) / 256, 256, 0, stream>>>(
      new_x, degp, xembed, xmaped, accp);
  // 5) labeled-node KL
  lbpr_kernel<<<(NLABELED + 127) / 128, 128, 0, stream>>>(
      xmaped, ylprob, clW, clb, idvert, accp);
  // 6) finalize 4 scalars
  finalize_kernel<<<1, 32, 0, stream>>>(accp, (float*)d_out);
}
